// DotProductAttention_25125558682349
// MI455X (gfx1250) — matrix operand order repaired and checked by reading, not compiled, not run on hardware
//
#include <hip/hip_runtime.h>
#include <hip/hip_bf16.h>

// Problem constants (match reference)
constexpr int BATCH = 32;
constexpr int QLEN  = 2048;
constexpr int KLEN  = 2048;
constexpr int DHEAD = 128;

typedef __attribute__((ext_vector_type(16))) __bf16 v16bf;
typedef __attribute__((ext_vector_type(8)))  float  v8f;

union Frag {
    uint4  q[2];
    v16bf  v;
};

__device__ __forceinline__ unsigned short f32_to_bf16(float f) {
    union { float f; unsigned u; } x; x.f = f;
    unsigned r = x.u + 0x7FFFu + ((x.u >> 16) & 1u);   // round-to-nearest-even
    return (unsigned short)(r >> 16);
}

// ---------------------------------------------------------------------------
// Kernel 1: Z = (Q * 1/sqrt(D)) @ K^T  -> written raw into A region of d_out
// Block: 128 threads (4 waves). Each block: 64 Q rows.
// Per iteration: 64-wide N tile -> 16 WMMAs per wave per barrier pair.
// ---------------------------------------------------------------------------
__global__ void __launch_bounds__(128)
attn_qk_kernel(const float* __restrict__ Qg, const float* __restrict__ Kg,
               float* __restrict__ outA, float scale) {
    constexpr int QSTR = DHEAD + 8;   // padded LDS row stride (halfwords)
    __shared__ alignas(16) unsigned short Qs[64 * QSTR];
    __shared__ alignas(16) unsigned short Ks[64 * QSTR];

    const int b    = blockIdx.x >> 5;           // 32 q-tiles per batch
    const int q0   = (blockIdx.x & 31) * 64;
    const int tid  = threadIdx.x;
    const int wave = tid >> 5;
    const int lane = tid & 31;
    const int m0   = wave * 16;

    // Stage Q tile (64 x 128) as bf16, pre-scaled.
    for (int idx = tid; idx < 64 * DHEAD; idx += 128) {
        const int m = idx >> 7, d = idx & 127;
        const float q = Qg[((size_t)b * QLEN + q0 + m) * DHEAD + d] * scale;
        Qs[m * QSTR + d] = f32_to_bf16(q);
    }

    const int rowA = m0 + (lane & 15);
    const int kbA  = (lane >> 4) * 8;    // A-operand K sub-base per half-wave
    const int rB   = lane & 15;          // B-operand: Ks row = local n
    const int dbB  = (lane >> 4) * 8;    // B-operand K sub-base per half-wave

    for (int nt0 = 0; nt0 < KLEN / 64; ++nt0) {
        __syncthreads();
        // Stage 64 K-rows (64 x 128) as bf16.
        const float* kPtr = Kg + ((size_t)b * KLEN + nt0 * 64) * DHEAD;
        for (int idx = tid; idx < 64 * DHEAD; idx += 128) {
            const int r = idx >> 7, d = idx & 127;
            Ks[r * QSTR + d] = f32_to_bf16(kPtr[(size_t)r * DHEAD + d]);
        }
        // Prefetch next 64-row K tile into cache while we compute.
        if (nt0 + 1 < KLEN / 64)
            __builtin_prefetch(kPtr + (size_t)64 * DHEAD + tid * 16, 0, 3);
        __syncthreads();

        v8f acc[4];
        #pragma unroll
        for (int s = 0; s < 4; ++s) acc[s] = (v8f){};

        #pragma unroll
        for (int ks = 0; ks < DHEAD; ks += 32) {
            Frag a;
            a.q[0] = *(const uint4*)&Qs[rowA * QSTR + ks + kbA];
            a.q[1] = *(const uint4*)&Qs[rowA * QSTR + ks + kbA + 16];
            #pragma unroll
            for (int s = 0; s < 4; ++s) {
                Frag bb;
                const int rs = s * 16 + rB;
                bb.q[0] = *(const uint4*)&Ks[rs * QSTR + ks + dbB];
                bb.q[1] = *(const uint4*)&Ks[rs * QSTR + ks + dbB + 16];
                acc[s] = __builtin_amdgcn_wmma_f32_16x16x32_bf16(
                             false, a.v, false, bb.v, (short)0, acc[s], false, false);
            }
        }

        // Store 4 x (16x16) score subtiles.
        #pragma unroll
        for (int s = 0; s < 4; ++s) {
            #pragma unroll
            for (int r = 0; r < 8; ++r) {
                const int m = m0 + r + ((lane >> 4) * 8);
                const int n = nt0 * 64 + s * 16 + (lane & 15);
                outA[((size_t)b * QLEN + q0 + m) * KLEN + n] = acc[s][r];
            }
        }
    }
}

// ---------------------------------------------------------------------------
// Kernel 2: in-place row softmax over A. One wave per row (2048 = 64/lane).
// Block: 256 threads = 8 rows.
// ---------------------------------------------------------------------------
__global__ void __launch_bounds__(256)
attn_softmax_kernel(float* __restrict__ A) {
    const int wave = threadIdx.x >> 5;
    const int lane = threadIdx.x & 31;
    const size_t row = (size_t)blockIdx.x * 8 + wave;
    float* __restrict__ p = A + row * KLEN;

    float v[64];
    float m = -3.0e38f;
    #pragma unroll
    for (int i = 0; i < 64; ++i) {
        v[i] = p[i * 32 + lane];
        m = fmaxf(m, v[i]);
    }
    #pragma unroll
    for (int off = 16; off >= 1; off >>= 1)
        m = fmaxf(m, __shfl_xor(m, off, 32));

    float s = 0.0f;
    #pragma unroll
    for (int i = 0; i < 64; ++i) {
        v[i] = __expf(v[i] - m);
        s += v[i];
    }
    #pragma unroll
    for (int off = 16; off >= 1; off >>= 1)
        s += __shfl_xor(s, off, 32);

    const float inv = 1.0f / s;
    #pragma unroll
    for (int i = 0; i < 64; ++i)
        p[i * 32 + lane] = v[i] * inv;
}

// ---------------------------------------------------------------------------
// Kernel 3: V_out = A @ V.  Block: 128 threads (4 waves), 64 q-rows per block,
// k-loop of 32 over the 2048 attention columns, 8 N-tiles (D=128).
// ---------------------------------------------------------------------------
__global__ void __launch_bounds__(128)
attn_av_kernel(const float* __restrict__ A, const float* __restrict__ Vg,
               float* __restrict__ outV) {
    constexpr int ASTR = 32 + 8;   // padded stride (halfwords)
    __shared__ alignas(16) unsigned short As[64  * ASTR];   // A tile, row-major
    __shared__ alignas(16) unsigned short Vs[DHEAD * ASTR]; // V tile, TRANSPOSED [d][k]

    const int b    = blockIdx.x >> 5;
    const int q0   = (blockIdx.x & 31) * 64;
    const int tid  = threadIdx.x;
    const int wave = tid >> 5;
    const int lane = tid & 31;
    const int m0   = wave * 16;

    const int rowA = m0 + (lane & 15);
    const int kbA  = (lane >> 4) * 8;
    const int kbB  = (lane >> 4) * 8;

    v8f acc[8];
    #pragma unroll
    for (int i = 0; i < 8; ++i) acc[i] = (v8f){};

    for (int k0 = 0; k0 < KLEN; k0 += 32) {
        __syncthreads();
        // Stage A tile (64 x 32) as bf16.
        const float* aPtr = A + ((size_t)b * QLEN + q0) * KLEN + k0;
        for (int idx = tid; idx < 64 * 32; idx += 128) {
            const int m = idx >> 5, k = idx & 31;
            As[m * ASTR + k] = f32_to_bf16(aPtr[(size_t)m * KLEN + k]);
        }
        // Stage V tile (32 x 128) transposed into [d][k].
        const float* vPtr = Vg + ((size_t)b * KLEN + k0) * DHEAD;
        for (int idx = tid; idx < 32 * DHEAD; idx += 128) {
            const int k = idx >> 7, d = idx & 127;
            Vs[d * ASTR + k] = f32_to_bf16(vPtr[(size_t)k * DHEAD + d]);
        }
        // Prefetch next tiles while this one is consumed.
        if (k0 + 32 < KLEN) {
            __builtin_prefetch(aPtr + 32 + (tid >> 2) * KLEN, 0, 3);
            __builtin_prefetch(vPtr + (size_t)32 * DHEAD + tid * 32, 0, 3);
        }
        __syncthreads();

        Frag a;
        a.q[0] = *(const uint4*)&As[rowA * ASTR + kbA];
        a.q[1] = *(const uint4*)&As[rowA * ASTR + kbA + 16];

        #pragma unroll
        for (int nt = 0; nt < 8; ++nt) {
            Frag bb;
            const int dv = nt * 16 + (lane & 15);
            bb.q[0] = *(const uint4*)&Vs[dv * ASTR + kbB];
            bb.q[1] = *(const uint4*)&Vs[dv * ASTR + kbB + 16];
            acc[nt] = __builtin_amdgcn_wmma_f32_16x16x32_bf16(
                          false, a.v, false, bb.v, (short)0, acc[nt], false, false);
        }
    }

    #pragma unroll
    for (int nt = 0; nt < 8; ++nt) {
        #pragma unroll
        for (int r = 0; r < 8; ++r) {
            const int m = m0 + r + ((lane >> 4) * 8);
            const int d = nt * 16 + (lane & 15);
            outV[((size_t)b * QLEN + q0 + m) * DHEAD + d] = acc[nt][r];
        }
    }
}

// ---------------------------------------------------------------------------
extern "C" void kernel_launch(void* const* d_in, const int* in_sizes, int n_in,
                              void* d_out, int out_size, void* d_ws, size_t ws_size,
                              hipStream_t stream) {
    const float* Qg = (const float*)d_in[0];
    const float* Kg = (const float*)d_in[1];
    const float* Vg = (const float*)d_in[2];

    float* outV = (float*)d_out;                                   // (B,Q,D)
    float* outA = (float*)d_out + (size_t)BATCH * QLEN * DHEAD;    // (B,Q,K)

    const float scale = 0.0883883476483184f;   // 1/sqrt(128)

    // 1) raw scores into A region
    attn_qk_kernel<<<dim3(BATCH * (QLEN / 64)), dim3(128), 0, stream>>>(
        Qg, Kg, outA, scale);
    // 2) in-place softmax over rows of A
    attn_softmax_kernel<<<dim3(BATCH * QLEN / 8), dim3(256), 0, stream>>>(outA);
    // 3) V_out = A @ V
    attn_av_kernel<<<dim3(BATCH * (QLEN / 64)), dim3(128), 0, stream>>>(
        outA, Vg, outV);
}
